// MyGRUCell_73383811219594
// MI455X (gfx1250) — matrix operand order repaired and checked by reading, not compiled, not run on hardware
//
#include <hip/hip_runtime.h>
#include <hip/hip_bf16.h>

typedef __attribute__((ext_vector_type(16))) _Float16 v16h;
typedef __attribute__((ext_vector_type(8)))  _Float16 v8h;
typedef __attribute__((ext_vector_type(8)))  float    v8f;
typedef __attribute__((ext_vector_type(4)))  float    v4f;

#define GRU_B 131072
#define GRU_D 256
#define GRU_H 128
#define GRU_O 256
#define WAVES_PER_BLOCK 4

// ---------------------------------------------------------------------------
// Weight prep: permute f32 row-major [K,N] into packed f16 WMMA B-fragments.
// Fragment (kt,nt) covers rows kt*32..kt*32+31, cols nt*16..nt*16+15.
// Within a fragment, lane l holds 16 halfs: h[j] = W[kt*32 + (l>>4)*8 + (j&7) + (j>>3)*16,
// nt*16 + (l&15)], stored contiguously so the GEMM kernel reads one v16h/lane.
// ---------------------------------------------------------------------------
__global__ void pack_b_kernel(const float* __restrict__ src,
                              _Float16* __restrict__ dst, int K, int N) {
  int t = blockIdx.x * 256 + threadIdx.x;
  if (t >= K * N) return;
  int numNt = N >> 4;
  int frag = t >> 9;          // 512 halfs per fragment
  int r    = t & 511;
  int lane = r >> 4;
  int j    = r & 15;
  int kt = frag / numNt;
  int nt = frag % numNt;
  int row = kt * 32 + ((lane >> 4) << 3) + (j & 7) + ((j >> 3) << 4);
  int col = nt * 16 + (lane & 15);
  dst[t] = (_Float16)src[row * N + col];
}

// ---------------------------------------------------------------------------
// Fused GRU cell kernel (one wave == one 16-row tile), gate math streamed
// per 16-column N-tile to keep accumulator pressure at 32 VGPRs.
// ---------------------------------------------------------------------------
__device__ __forceinline__ v8f wmma_f16(v16h a, v16h b, v8f c) {
  return __builtin_amdgcn_wmma_f32_16x16x32_f16(
      false, a, false, b, (short)0, c, false, false);
}

__device__ __forceinline__ v16h load_bfrag(const _Float16* __restrict__ p,
                                           int frag, int lane) {
  return *(const v16h*)(p + frag * 512 + lane * 16);
}

__device__ __forceinline__ float sigmoidf(float x) {
  return 1.0f / (1.0f + __expf(-x));
}

// Load one 16x32 f16 A-fragment from an f32 row-major matrix with ldc cols.
__device__ __forceinline__ v16h load_afrag_f32(const float* __restrict__ row,
                                               int kt, int kgrp) {
  int k0 = kt * 32 + kgrp * 8;
  v4f a0 = *(const v4f*)(row + k0);
  v4f a1 = *(const v4f*)(row + k0 + 4);
  v4f a2 = *(const v4f*)(row + k0 + 16);
  v4f a3 = *(const v4f*)(row + k0 + 20);
  v16h out;
#pragma unroll
  for (int j = 0; j < 4; ++j) {
    out[j]      = (_Float16)a0[j];
    out[4 + j]  = (_Float16)a1[j];
    out[8 + j]  = (_Float16)a2[j];
    out[12 + j] = (_Float16)a3[j];
  }
  return out;
}

__global__ __launch_bounds__(32 * WAVES_PER_BLOCK, 4)
void gru_fused_kernel(const float* __restrict__ X,
                      const float* __restrict__ Hst,
                      const _Float16* __restrict__ wz_p,
                      const _Float16* __restrict__ hz_p,
                      const _Float16* __restrict__ wr_p,
                      const _Float16* __restrict__ hr_p,
                      const _Float16* __restrict__ wmc_p,
                      const _Float16* __restrict__ hmc_p,
                      const _Float16* __restrict__ wy_p,
                      float* __restrict__ out,
                      float* __restrict__ next_out) {
  __shared__ _Float16 lds_next[WAVES_PER_BLOCK][16 * GRU_H];

  const int lane = threadIdx.x & 31;
  const int wave = threadIdx.x >> 5;
  const int tile = blockIdx.x * WAVES_PER_BLOCK + wave;
  const int row0 = tile * 16;

  const int l15  = lane & 15;   // A: row M;  B/C: col N
  const int kgrp = lane >> 4;   // 0 or 1

  // ---- load X tile as 8 A-fragments, h tile as 4 A-fragments (f32->f16) ---
  v16h xf[8];
  {
    const float* xrow = X + (size_t)(row0 + l15) * GRU_D;
#pragma unroll
    for (int kt = 0; kt < 8; ++kt) xf[kt] = load_afrag_f32(xrow, kt, kgrp);
  }
  v16h hf[4];
  {
    const float* hrow = Hst + (size_t)(row0 + l15) * GRU_H;
#pragma unroll
    for (int kt = 0; kt < 4; ++kt) hf[kt] = load_afrag_f32(hrow, kt, kgrp);
  }

  // ---- streamed gate GEMMs + elementwise, one 16-col N-tile at a time -----
  _Float16* nf = lds_next[wave];
  for (int nt = 0; nt < 8; ++nt) {
    v8f az = {}, ar = {}, am1 = {}, am2 = {};
#pragma unroll
    for (int kt = 0; kt < 8; ++kt) {
      az  = wmma_f16(xf[kt], load_bfrag(wz_p,  kt * 8 + nt, lane), az);
      ar  = wmma_f16(xf[kt], load_bfrag(wr_p,  kt * 8 + nt, lane), ar);
      am1 = wmma_f16(xf[kt], load_bfrag(wmc_p, kt * 8 + nt, lane), am1);
    }
#pragma unroll
    for (int kt = 0; kt < 4; ++kt) {
      az  = wmma_f16(hf[kt], load_bfrag(hz_p,  kt * 8 + nt, lane), az);
      ar  = wmma_f16(hf[kt], load_bfrag(hr_p,  kt * 8 + nt, lane), ar);
      am2 = wmma_f16(hf[kt], load_bfrag(hmc_p, kt * 8 + nt, lane), am2);
    }
    // C layout: reg r, lane grp 0: (M=r, N=l15); lane grp 1: (M=r+8, N=l15)
    int col = nt * 16 + l15;
#pragma unroll
    for (int r = 0; r < 8; ++r) {
      int m    = r + kgrp * 8;
      float z  = sigmoidf(az[r]);
      float rt = sigmoidf(ar[r]);
      float hc = sigmoidf(am1[r] + rt * am2[r]);
      float hv = Hst[(size_t)(row0 + m) * GRU_H + col];
      float ns = z * hv + (1.0f - z) * hc;
      next_out[(size_t)(row0 + m) * GRU_H + col] = ns;
      nf[m * GRU_H + col] = (_Float16)ns;
    }
  }
  __syncthreads();

  // ---- re-read next_state from LDS in A-fragment layout -------------------
  v16h nxf[4];
  {
    const _Float16* nrow = nf + l15 * GRU_H;
#pragma unroll
    for (int kt = 0; kt < 4; ++kt) {
      int k0 = kt * 32 + kgrp * 8;
      v8h lo = *(const v8h*)(nrow + k0);
      v8h hi = *(const v8h*)(nrow + k0 + 16);
#pragma unroll
      for (int j = 0; j < 8; ++j) {
        nxf[kt][j]     = lo[j];
        nxf[kt][8 + j] = hi[j];
      }
    }
  }

  // ---- output GEMM: out = next_state @ Wy  [16 x 256] ---------------------
  for (int nt = 0; nt < 16; ++nt) {
    v8f acc = {};
#pragma unroll
    for (int kt = 0; kt < 4; ++kt)
      acc = wmma_f16(nxf[kt], load_bfrag(wy_p, kt * 16 + nt, lane), acc);
    int col = nt * 16 + l15;
#pragma unroll
    for (int r = 0; r < 8; ++r) {
      int m = r + kgrp * 8;
      out[(size_t)(row0 + m) * GRU_O + col] = acc[r];
    }
  }
}

// ---------------------------------------------------------------------------
extern "C" void kernel_launch(void* const* d_in, const int* in_sizes, int n_in,
                              void* d_out, int out_size, void* d_ws, size_t ws_size,
                              hipStream_t stream) {
  const float* X    = (const float*)d_in[0];
  const float* h    = (const float*)d_in[1];
  const float* w_z  = (const float*)d_in[2];
  const float* h_z  = (const float*)d_in[3];
  const float* w_r  = (const float*)d_in[4];
  const float* h_r  = (const float*)d_in[5];
  const float* w_mc = (const float*)d_in[6];
  const float* h_mc = (const float*)d_in[7];
  const float* w_y  = (const float*)d_in[8];

  // packed f16 weight fragments in workspace (total 180224 halfs = 352 KB)
  _Float16* wz_p  = (_Float16*)d_ws;
  _Float16* wr_p  = wz_p  + GRU_D * GRU_H;   // 32768
  _Float16* wmc_p = wr_p  + GRU_D * GRU_H;
  _Float16* hz_p  = wmc_p + GRU_D * GRU_H;
  _Float16* hr_p  = hz_p  + GRU_H * GRU_H;   // 16384
  _Float16* hmc_p = hr_p  + GRU_H * GRU_H;
  _Float16* wy_p  = hmc_p + GRU_H * GRU_H;   // 32768

  const int nDH = GRU_D * GRU_H, nHH = GRU_H * GRU_H, nHO = GRU_H * GRU_O;
  pack_b_kernel<<<(nDH + 255) / 256, 256, 0, stream>>>(w_z,  wz_p,  GRU_D, GRU_H);
  pack_b_kernel<<<(nDH + 255) / 256, 256, 0, stream>>>(w_r,  wr_p,  GRU_D, GRU_H);
  pack_b_kernel<<<(nDH + 255) / 256, 256, 0, stream>>>(w_mc, wmc_p, GRU_D, GRU_H);
  pack_b_kernel<<<(nHH + 255) / 256, 256, 0, stream>>>(h_z,  hz_p,  GRU_H, GRU_H);
  pack_b_kernel<<<(nHH + 255) / 256, 256, 0, stream>>>(h_r,  hr_p,  GRU_H, GRU_H);
  pack_b_kernel<<<(nHH + 255) / 256, 256, 0, stream>>>(h_mc, hmc_p, GRU_H, GRU_H);
  pack_b_kernel<<<(nHO + 255) / 256, 256, 0, stream>>>(w_y,  wy_p,  GRU_H, GRU_O);

  float* out      = (float*)d_out;                       // [B, 256]
  float* next_out = out + (size_t)GRU_B * GRU_O;         // [B, 128]

  const int blocks = GRU_B / (16 * WAVES_PER_BLOCK);     // 2048
  gru_fused_kernel<<<blocks, 32 * WAVES_PER_BLOCK, 0, stream>>>(
      X, h, wz_p, hz_p, wr_p, hr_p, wmc_p, hmc_p, wy_p, out, next_out);
}
